// MLVFusion_5480378270377
// MI455X (gfx1250) — matrix operand order repaired and checked by reading, not compiled, not run on hardware
//
#include <hip/hip_runtime.h>
#include <hip/hip_bf16.h>

// ---------------------------------------------------------------------------
// MLVFusion for MI455X (gfx1250): bf16 WMMA GEMMs + TDM async tile staging.
// ---------------------------------------------------------------------------

#define BATCH   2
#define DMODEL  768
#define NHEADS  16
#define DHEAD   48
#define LQ      9216      // 96*96
#define LIN     14592     // 256+1024+4096+9216

typedef float  v8f   __attribute__((ext_vector_type(8)));
typedef __bf16 v16bf __attribute__((ext_vector_type(16)));
typedef __bf16 v8bf  __attribute__((ext_vector_type(8)));
typedef unsigned int u32x4 __attribute__((ext_vector_type(4)));
typedef int          i32x4 __attribute__((ext_vector_type(4)));
typedef int          i32x8 __attribute__((ext_vector_type(8)));

// ---------------------------------------------------------------------------
// TDM: load a (tileH x 32) bf16 tile (row stride = strideElems elements in
// global) into LDS with a 4-dword pad every 16 dwords (=> LDS row stride 20
// dwords = 80 B, keeps rows 16B-aligned for ds_load_b128 and conflict-free).
// Descriptor packed per CDNA5 ISA ch. 8 (async tensor).
// ---------------------------------------------------------------------------
__device__ __forceinline__ void tdm_load_tile32(unsigned ldsAddr,
                                                const void* gptr,
                                                int tileH, int strideElems) {
  unsigned long long ga = (unsigned long long)gptr;
  // group0: count=1 | lds_addr | global_addr[56:0] | type=2
  u32x4 g0 = { 1u,
               ldsAddr,
               (unsigned)(ga & 0xFFFFFFFFull),
               (unsigned)((ga >> 32) & 0x01FFFFFFull) | (2u << 30) };
  const int tileW = 32;                       // elements (bf16) per tile row
  // group1 dword0: data_size=1(2B)<<16 | pad_en<<20 | pad_interval=3(16dw)<<22
  //                | pad_amount=3(4dw)<<25
  int d0 = (1 << 16) | (1 << 20) | (3 << 22) | (3 << 25);
  int d1 = (tileW & 0xFFFF) << 16;                              // tensor_dim0 lo
  int d2 = ((tileW >> 16) & 0xFFFF) | ((tileH & 0xFFFF) << 16); // td0 hi | td1 lo
  int d3 = ((tileH >> 16) & 0xFFFF) | ((tileW & 0xFFFF) << 16); // td1 hi | tile_dim0
  int d4 = (tileH & 0xFFFF);                                    // tile_dim1 | tile_dim2=0
  int d5 = strideElems;                                         // tensor_dim0_stride lo32
  i32x8 g1 = { d0, d1, d2, d3, d4, d5, 0, 0 };
  i32x4 z4 = { 0, 0, 0, 0 };
#if defined(__clang_major__) && (__clang_major__ >= 23)
  i32x8 z8 = { 0, 0, 0, 0, 0, 0, 0, 0 };
  __builtin_amdgcn_tensor_load_to_lds(g0, g1, z4, z4, z8, 0);
#else
  __builtin_amdgcn_tensor_load_to_lds(g0, g1, z4, z4, 0);
#endif
}

// Two 16B LDS loads -> one 16-lane bf16 fragment.
__device__ __forceinline__ v16bf ldfrag(const unsigned* b0, const unsigned* b1) {
  uint4 u0 = *(const uint4*)b0;
  uint4 u1 = *(const uint4*)b1;
  v8bf h0 = __builtin_bit_cast(v8bf, u0);
  v8bf h1 = __builtin_bit_cast(v8bf, u1);
  return __builtin_shufflevector(h0, h1, 0, 1, 2, 3, 4, 5, 6, 7,
                                 8, 9, 10, 11, 12, 13, 14, 15);
}

// ---------------------------------------------------------------------------
// GEMM: C(MxN) = A(MxK bf16, row major) * Wt(NxK bf16, row major => B=W K x N)
//       + bias.  BM=128 BN=64 BK=32, 256 threads = 8 waves, wave = 32x32 tile.
// mode 0: store f32   mode 1: store bf16
// mode 2: out = query + g1*(attn + g2*acc)   (final fused epilogue, f32 out)
// ---------------------------------------------------------------------------
#define BM 128
#define BN 64
#define BK 32
#define LDT 20   // dwords per LDS tile row (16 data + 4 pad)

__global__ __launch_bounds__(256) void gemm_wmma_bf16(
    const __hip_bfloat16* __restrict__ A, const __hip_bfloat16* __restrict__ Wt,
    const float* __restrict__ bias, int M, int N, int K, int mode,
    void* __restrict__ outp,
    const float* __restrict__ attn, const float* __restrict__ query,
    const float* __restrict__ gma1, const float* __restrict__ gma2) {
  __shared__ __align__(16) unsigned shA[2][BM * LDT];
  __shared__ __align__(16) unsigned shB[2][BN * LDT];

  const int bm = blockIdx.y, bn = blockIdx.x;
  const int tid  = threadIdx.x;
  const int lane = tid & 31;
  const int wv   = tid >> 5;      // 0..7
  const int wm   = wv >> 1;       // 0..3 -> M offset 32*wm
  const int wn   = wv & 1;        // 0..1 -> N offset 32*wn
  const int half = lane >> 4;
  const int l15  = lane & 15;
  const int nk   = K / BK;

  // Opaque dead store: prevents "LDS object is never written => loads undef"
  // reasoning (TDM writes are invisible to the compiler).
  unsigned zr = 0;
  asm("" : "+s"(zr));
  if (zr) { shA[0][0] = zr; shB[0][0] = zr; }

  v8f acc[2][2] = {};

  const size_t arow = (size_t)bm * BM;
  const size_t brow = (size_t)bn * BN;

  if (wv == 0) {
    tdm_load_tile32((unsigned)(size_t)&shA[0][0], A + arow * K, BM, K);
    tdm_load_tile32((unsigned)(size_t)&shB[0][0], Wt + brow * K, BN, K);
  }

#pragma unroll 2
  for (int kt = 0; kt < nk; ++kt) {
    const int cur = kt & 1;
    if (wv == 0) {
      if (kt + 1 < nk) {
        const int nxt = cur ^ 1;
        tdm_load_tile32((unsigned)(size_t)&shA[nxt][0],
                        A + arow * K + (size_t)(kt + 1) * BK, BM, K);
        tdm_load_tile32((unsigned)(size_t)&shB[nxt][0],
                        Wt + brow * K + (size_t)(kt + 1) * BK, BN, K);
        __builtin_amdgcn_s_wait_tensorcnt(2);   // current pair has landed
      } else {
        __builtin_amdgcn_s_wait_tensorcnt(0);
      }
    }
    __syncthreads();

    const unsigned* pa = &shA[cur][0];
    const unsigned* pb = &shB[cur][0];

    // A frag i: dwords [row*LDT + 4*half + 0..3] and [+8 .. +11]
    v16bf afr[2], bfr[2];
#pragma unroll
    for (int i = 0; i < 2; ++i) {
      const int row = wm * 32 + i * 16 + l15;
      const unsigned* b0 = pa + row * LDT + 4 * half;
      afr[i] = ldfrag(b0, b0 + 8);
    }
    // B frag j: dwords [col*LDT + 4*half + 0..3] and [+8 .. +11]
#pragma unroll
    for (int j = 0; j < 2; ++j) {
      const int col = wn * 32 + j * 16 + l15;
      const unsigned* b0 = pb + col * LDT + 4 * half;
      bfr[j] = ldfrag(b0, b0 + 8);
    }

    acc[0][0] = __builtin_amdgcn_wmma_f32_16x16x32_bf16(false, afr[0], false, bfr[0], (short)0, acc[0][0], false, false);
    acc[0][1] = __builtin_amdgcn_wmma_f32_16x16x32_bf16(false, afr[0], false, bfr[1], (short)0, acc[0][1], false, false);
    acc[1][0] = __builtin_amdgcn_wmma_f32_16x16x32_bf16(false, afr[1], false, bfr[0], (short)0, acc[1][0], false, false);
    acc[1][1] = __builtin_amdgcn_wmma_f32_16x16x32_bf16(false, afr[1], false, bfr[1], (short)0, acc[1][1], false, false);

    __syncthreads();  // compute done before next TDM overwrites this buffer
  }

  // Epilogue.  C/D layout: VGPR r -> M = r + 8*half, N = lane&15.
#pragma unroll
  for (int i = 0; i < 2; ++i) {
#pragma unroll
    for (int j = 0; j < 2; ++j) {
      const int col = bn * BN + wn * 32 + j * 16 + l15;
      const float bv  = bias[col];
      const float g1c = (mode == 2) ? gma1[col] : 0.f;
      const float g2c = (mode == 2) ? gma2[col] : 0.f;
#pragma unroll
      for (int r = 0; r < 8; ++r) {
        const int row = bm * BM + wm * 32 + i * 16 + half * 8 + r;
        const float v = acc[i][j][r] + bv;
        const size_t idx = (size_t)row * (size_t)N + (size_t)col;
        if (mode == 0) {
          ((float*)outp)[idx] = v;
        } else if (mode == 1) {
          ((__hip_bfloat16*)outp)[idx] = __float2bfloat16(v);
        } else {
          ((float*)outp)[idx] = query[idx] + g1c * (attn[idx] + g2c * v);
        }
      }
    }
  }
}

// ---------------------------------------------------------------------------
// LayerNorm(768) -> bf16, one wave (32 lanes) per row, 8 rows / block.
// Row remap lets us scatter src0..src3 into the concatenated feat buffer.
// ---------------------------------------------------------------------------
__global__ __launch_bounds__(256) void ln_to_bf16(
    const float* __restrict__ x, __hip_bfloat16* __restrict__ y,
    const float* __restrict__ g, const float* __restrict__ b,
    int rowsTotal, int rowsPerBatchSrc, int dstRowsPerBatch, int dstRowOff) {
  const int row = blockIdx.x * 8 + (threadIdx.x >> 5);
  if (row >= rowsTotal) return;
  const int lane = threadIdx.x & 31;
  const float* xr = x + (size_t)row * DMODEL;

  float v[24];
  float s = 0.f;
#pragma unroll
  for (int i = 0; i < 24; ++i) { v[i] = xr[lane + 32 * i]; s += v[i]; }
#pragma unroll
  for (int o = 16; o >= 1; o >>= 1) s += __shfl_xor(s, o, 32);
  const float mean = s * (1.f / 768.f);

  float q = 0.f;
#pragma unroll
  for (int i = 0; i < 24; ++i) { const float d = v[i] - mean; q += d * d; }
#pragma unroll
  for (int o = 16; o >= 1; o >>= 1) q += __shfl_xor(q, o, 32);
  const float rs = rsqrtf(q * (1.f / 768.f) + 1e-6f);

  const int bi = row / rowsPerBatchSrc;
  const int r  = row - bi * rowsPerBatchSrc;
  __hip_bfloat16* yr =
      y + ((size_t)bi * dstRowsPerBatch + dstRowOff + r) * DMODEL;
#pragma unroll
  for (int i = 0; i < 24; ++i) {
    const int c = lane + 32 * i;
    yr[c] = __float2bfloat16((v[i] - mean) * rs * g[c] + b[c]);
  }
}

// fp32 (KxN) weight -> bf16 transposed (NxK) so GEMM B tiles are row slabs.
__global__ __launch_bounds__(256) void wconv_transpose(
    const float* __restrict__ w, __hip_bfloat16* __restrict__ wt, int K, int N) {
  const size_t id = (size_t)blockIdx.x * 256 + threadIdx.x;
  if (id >= (size_t)K * N) return;
  const int n = (int)(id % N);
  const int k = (int)(id / N);
  wt[(size_t)n * K + k] = __float2bfloat16(w[id]);
}

// ---------------------------------------------------------------------------
// Deformable sampling: softmax over L*P logits, bilinear gather of bf16 value,
// weighted sum -> bf16 samp.  One thread per (b,q,h,d).
// ---------------------------------------------------------------------------
__global__ __launch_bounds__(256) void msda_sample(
    const __hip_bfloat16* __restrict__ value, const float* __restrict__ off,
    const float* __restrict__ logit, __hip_bfloat16* __restrict__ out,
    int Lin, int nLev,
    int H0, int W0, int S0, int H1, int W1, int S1,
    int H2, int W2, int S2, int H3, int W3, int S3) {
  const int total = BATCH * LQ * NHEADS * DHEAD;
  const int id = blockIdx.x * 256 + threadIdx.x;
  if (id >= total) return;
  const int d = id % DHEAD;
  const int h = (id / DHEAD) & (NHEADS - 1);
  const int q = (id / (DHEAD * NHEADS)) % LQ;
  const int b = id / (DHEAD * NHEADS * LQ);
  const int LP = nLev * 4;

  // softmax over L*P attention logits for this (b,q,h)
  const float* lg = logit + ((size_t)b * LQ + q) * (size_t)(NHEADS * LP) + h * LP;
  float mx = -1e30f;
  for (int i = 0; i < LP; ++i) mx = fmaxf(mx, lg[i]);
  float wts[16];
  float se = 0.f;
  for (int i = 0; i < LP; ++i) { wts[i] = __expf(lg[i] - mx); se += wts[i]; }
  const float inv = 1.f / se;

  const float rx = ((float)(q % 96) + 0.5f) * (1.f / 96.f);
  const float ry = ((float)(q / 96) + 0.5f) * (1.f / 96.f);
  const float* ob =
      off + ((size_t)b * LQ + q) * (size_t)(NHEADS * LP * 2) + h * LP * 2;

  const int Hs[4] = { H0, H1, H2, H3 };
  const int Ws[4] = { W0, W1, W2, W3 };
  const int Ss[4] = { S0, S1, S2, S3 };

  float acc = 0.f;
  for (int l = 0; l < nLev; ++l) {
    const int Hh = Hs[l], Ww = Ws[l], st = Ss[l];
    const __hip_bfloat16* vb =
        value + ((size_t)b * Lin + st) * DMODEL + h * DHEAD + d;
    for (int p = 0; p < 4; ++p) {
      const float ox = ob[(l * 4 + p) * 2 + 0];
      const float oy = ob[(l * 4 + p) * 2 + 1];
      const float px = (rx + ox / (float)Ww) * (float)Ww - 0.5f;
      const float py = (ry + oy / (float)Hh) * (float)Hh - 0.5f;
      const float x0f = floorf(px), y0f = floorf(py);
      const float lx = px - x0f, ly = py - y0f;
      const int x0 = (int)x0f, y0 = (int)y0f;
      float s = 0.f;
      {
        const int xi = x0, yi = y0; const float wc = (1.f - lx) * (1.f - ly);
        if (xi >= 0 && xi < Ww && yi >= 0 && yi < Hh)
          s += wc * __bfloat162float(vb[((size_t)(yi * Ww + xi)) * DMODEL]);
      }
      {
        const int xi = x0 + 1, yi = y0; const float wc = lx * (1.f - ly);
        if (xi >= 0 && xi < Ww && yi >= 0 && yi < Hh)
          s += wc * __bfloat162float(vb[((size_t)(yi * Ww + xi)) * DMODEL]);
      }
      {
        const int xi = x0, yi = y0 + 1; const float wc = (1.f - lx) * ly;
        if (xi >= 0 && xi < Ww && yi >= 0 && yi < Hh)
          s += wc * __bfloat162float(vb[((size_t)(yi * Ww + xi)) * DMODEL]);
      }
      {
        const int xi = x0 + 1, yi = y0 + 1; const float wc = lx * ly;
        if (xi >= 0 && xi < Ww && yi >= 0 && yi < Hh)
          s += wc * __bfloat162float(vb[((size_t)(yi * Ww + xi)) * DMODEL]);
      }
      acc += (wts[l * 4 + p] * inv) * s;
    }
  }
  out[((size_t)b * LQ + q) * DMODEL + h * DHEAD + d] = __float2bfloat16(acc);
}

// ---------------------------------------------------------------------------
// Host orchestration
// ---------------------------------------------------------------------------
extern "C" void kernel_launch(void* const* d_in, const int* in_sizes, int n_in,
                              void* d_out, int out_size, void* d_ws, size_t ws_size,
                              hipStream_t stream) {
  (void)in_sizes; (void)n_in; (void)out_size; (void)ws_size;
  const float* src0   = (const float*)d_in[0];
  const float* src1   = (const float*)d_in[1];
  const float* src2   = (const float*)d_in[2];
  const float* src3   = (const float*)d_in[3];
  const float* qn_g   = (const float*)d_in[4];
  const float* qn_b   = (const float*)d_in[5];
  const float* fn_g   = (const float*)d_in[6];
  const float* fn_b   = (const float*)d_in[7];
  const float* n1_g   = (const float*)d_in[8];
  const float* n1_b   = (const float*)d_in[9];
  const float* gamma1 = (const float*)d_in[10];
  const float* gamma2 = (const float*)d_in[11];
  const float* c_Wv   = (const float*)d_in[12];
  const float* c_bv   = (const float*)d_in[13];
  const float* c_Woff = (const float*)d_in[14];
  const float* c_boff = (const float*)d_in[15];
  const float* c_Wa   = (const float*)d_in[16];
  const float* c_ba   = (const float*)d_in[17];
  const float* c_Wo   = (const float*)d_in[18];
  const float* c_bo   = (const float*)d_in[19];
  const float* s_Wv   = (const float*)d_in[20];
  const float* s_bv   = (const float*)d_in[21];
  const float* s_Woff = (const float*)d_in[22];
  const float* s_boff = (const float*)d_in[23];
  const float* s_Wa   = (const float*)d_in[24];
  const float* s_ba   = (const float*)d_in[25];
  const float* s_Wo   = (const float*)d_in[26];
  const float* s_bo   = (const float*)d_in[27];

  const int KD = DMODEL;                 // 768
  const int Mq = BATCH * LQ;             // 18432
  const int Mv = BATCH * LIN;            // 29184

  size_t cur = 0;
  auto wsAlloc = [&](size_t bytes) -> void* {
    void* p = (char*)d_ws + cur;
    cur += (bytes + 255) & ~(size_t)255;
    return p;
  };

  // bf16 transposed weights (N x K)
  __hip_bfloat16* tWv_c  = (__hip_bfloat16*)wsAlloc((size_t)768 * KD * 2);
  __hip_bfloat16* tWoff_c= (__hip_bfloat16*)wsAlloc((size_t)512 * KD * 2);
  __hip_bfloat16* tWa_c  = (__hip_bfloat16*)wsAlloc((size_t)256 * KD * 2);
  __hip_bfloat16* tWo_c  = (__hip_bfloat16*)wsAlloc((size_t)768 * KD * 2);
  __hip_bfloat16* tWv_s  = (__hip_bfloat16*)wsAlloc((size_t)768 * KD * 2);
  __hip_bfloat16* tWoff_s= (__hip_bfloat16*)wsAlloc((size_t)128 * KD * 2);
  __hip_bfloat16* tWa_s  = (__hip_bfloat16*)wsAlloc((size_t)64  * KD * 2);
  __hip_bfloat16* tWo_s  = (__hip_bfloat16*)wsAlloc((size_t)768 * KD * 2);

  // activations
  __hip_bfloat16* qn    = (__hip_bfloat16*)wsAlloc((size_t)Mq * KD * 2); // LN(query) bf16; reused as samp2
  __hip_bfloat16* fn    = (__hip_bfloat16*)wsAlloc((size_t)Mv * KD * 2); // LN(feat) bf16; reused as value2
  __hip_bfloat16* value = (__hip_bfloat16*)wsAlloc((size_t)Mv * KD * 2);
  float*          off1  = (float*)wsAlloc((size_t)Mq * 512 * 4);         // reused as off2
  float*          wlog1 = (float*)wsAlloc((size_t)Mq * 256 * 4);         // reused as wlog2
  __hip_bfloat16* samp1 = (__hip_bfloat16*)wsAlloc((size_t)Mq * KD * 2);
  float*          attnf = (float*)wsAlloc((size_t)Mq * KD * 4);
  __hip_bfloat16* attn1 = (__hip_bfloat16*)wsAlloc((size_t)Mq * KD * 2);
  __hip_bfloat16* value2 = fn;
  float*          off2   = off1;
  float*          wlog2  = wlog1;
  __hip_bfloat16* samp2  = qn;

  auto wgrid = [](int K, int N) { return dim3((unsigned)(((size_t)K * N + 255) / 256)); };
  // 1) weight convert + transpose
  wconv_transpose<<<wgrid(KD, 768), 256, 0, stream>>>(c_Wv,   tWv_c,   KD, 768);
  wconv_transpose<<<wgrid(KD, 512), 256, 0, stream>>>(c_Woff, tWoff_c, KD, 512);
  wconv_transpose<<<wgrid(KD, 256), 256, 0, stream>>>(c_Wa,   tWa_c,   KD, 256);
  wconv_transpose<<<wgrid(KD, 768), 256, 0, stream>>>(c_Wo,   tWo_c,   KD, 768);
  wconv_transpose<<<wgrid(KD, 768), 256, 0, stream>>>(s_Wv,   tWv_s,   KD, 768);
  wconv_transpose<<<wgrid(KD, 128), 256, 0, stream>>>(s_Woff, tWoff_s, KD, 128);
  wconv_transpose<<<wgrid(KD, 64),  256, 0, stream>>>(s_Wa,   tWa_s,   KD, 64);
  wconv_transpose<<<wgrid(KD, 768), 256, 0, stream>>>(s_Wo,   tWo_s,   KD, 768);

  // 2) layernorms -> bf16
  ln_to_bf16<<<dim3((Mq + 7) / 8), 256, 0, stream>>>(src3, qn, qn_g, qn_b, Mq, LQ, LQ, 0);
  ln_to_bf16<<<dim3((BATCH * 256  + 7) / 8), 256, 0, stream>>>(src0, fn, fn_g, fn_b, BATCH * 256,  256,  LIN, 0);
  ln_to_bf16<<<dim3((BATCH * 1024 + 7) / 8), 256, 0, stream>>>(src1, fn, fn_g, fn_b, BATCH * 1024, 1024, LIN, 256);
  ln_to_bf16<<<dim3((BATCH * 4096 + 7) / 8), 256, 0, stream>>>(src2, fn, fn_g, fn_b, BATCH * 4096, 4096, LIN, 1280);
  ln_to_bf16<<<dim3((BATCH * LQ   + 7) / 8), 256, 0, stream>>>(src3, fn, fn_g, fn_b, BATCH * LQ,   LQ,   LIN, 5376);

  // 3) stage-1 projections
  gemm_wmma_bf16<<<dim3(768 / BN, Mv / BM), 256, 0, stream>>>(fn, tWv_c,   c_bv,   Mv, 768, KD, 1, value, nullptr, nullptr, nullptr, nullptr);
  gemm_wmma_bf16<<<dim3(512 / BN, Mq / BM), 256, 0, stream>>>(qn, tWoff_c, c_boff, Mq, 512, KD, 0, off1,  nullptr, nullptr, nullptr, nullptr);
  gemm_wmma_bf16<<<dim3(256 / BN, Mq / BM), 256, 0, stream>>>(qn, tWa_c,   c_ba,   Mq, 256, KD, 0, wlog1, nullptr, nullptr, nullptr, nullptr);

  // 4) stage-1 deformable sampling
  const int sampThreads = BATCH * LQ * NHEADS * DHEAD;
  msda_sample<<<dim3((sampThreads + 255) / 256), 256, 0, stream>>>(
      value, off1, wlog1, samp1, LIN, 4,
      16, 16, 0, 32, 32, 256, 64, 64, 1280, 96, 96, 5376);

  // 5) attn = samp1 @ c_Wo + c_bo (f32), attn1 = LN(attn) bf16
  gemm_wmma_bf16<<<dim3(768 / BN, Mq / BM), 256, 0, stream>>>(samp1, tWo_c, c_bo, Mq, 768, KD, 0, attnf, nullptr, nullptr, nullptr, nullptr);
  ln_to_bf16<<<dim3((Mq + 7) / 8), 256, 0, stream>>>(attnf, attn1, n1_g, n1_b, Mq, LQ, LQ, 0);

  // 6) stage-2 projections (query = feat = attn1)
  gemm_wmma_bf16<<<dim3(768 / BN, Mq / BM), 256, 0, stream>>>(attn1, tWv_s,   s_bv,   Mq, 768, KD, 1, value2, nullptr, nullptr, nullptr, nullptr);
  gemm_wmma_bf16<<<dim3(128 / BN, Mq / BM), 256, 0, stream>>>(attn1, tWoff_s, s_boff, Mq, 128, KD, 0, off2,   nullptr, nullptr, nullptr, nullptr);
  gemm_wmma_bf16<<<dim3(64  / BN, Mq / BM), 256, 0, stream>>>(attn1, tWa_s,   s_ba,   Mq, 64,  KD, 0, wlog2,  nullptr, nullptr, nullptr, nullptr);

  // 7) stage-2 sampling (single 96x96 level)
  msda_sample<<<dim3((sampThreads + 255) / 256), 256, 0, stream>>>(
      value2, off2, wlog2, samp2, LQ, 1,
      96, 96, 0, 1, 1, 0, 1, 1, 0, 1, 1, 0);

  // 8) final fused GEMM: out = query + g1*(attn + g2*(samp2 @ s_Wo + s_bo))
  gemm_wmma_bf16<<<dim3(768 / BN, Mq / BM), 256, 0, stream>>>(
      samp2, tWo_s, s_bo, Mq, 768, KD, 2, d_out, attnf, src3, gamma1, gamma2);
}
